// RnnLayer_24747601559644
// MI455X (gfx1250) — compile-verified, matrix operand order checked by reading, not run on hardware
//
#include <hip/hip_runtime.h>
#include <hip/hip_bf16.h>

// ---------------- problem constants ----------------
#define Bsz 16
#define Tsz 4096
#define Dsz 256
#define Lsz 512
#define TB  (Tsz * Bsz)          // 65536 time-major rows (row = t*B + b)
#define NWG 16                   // workgroups cooperating on the recurrence
#define LDS_PITCH 520            // 512 + 8 bf16 pad -> conflict-free b128 LDS reads

typedef __attribute__((ext_vector_type(16))) __bf16 v16bf;
typedef __attribute__((ext_vector_type(8)))  __bf16 v8bf;
typedef __attribute__((ext_vector_type(8)))  float  v8f;

static __device__ __forceinline__ float bf2f(__bf16 b) {
    unsigned short s = __builtin_bit_cast(unsigned short, b);
    unsigned u = ((unsigned)s) << 16;
    return __builtin_bit_cast(float, u);
}
static __device__ __forceinline__ __bf16 f2bf(float f) {
    unsigned u = __builtin_bit_cast(unsigned, f);
    u += 0x7fffu + ((u >> 16) & 1u);          // round-to-nearest-even
    unsigned short s = (unsigned short)(u >> 16);
    return __builtin_bit_cast(__bf16, s);
}
static __device__ __forceinline__ v16bf cat16(v8bf lo, v8bf hi) {
    return __builtin_shufflevector(lo, hi, 0,1,2,3,4,5,6,7,8,9,10,11,12,13,14,15);
}

// ---------------- phase 0: weight conversion + state init ----------------
__global__ void __launch_bounds__(256)
prep_kernel(const float* __restrict__ Wi, const float* __restrict__ Wm,
            __bf16* __restrict__ Wibf, __bf16* __restrict__ Wmbf,
            __bf16* __restrict__ hbuf, unsigned* __restrict__ bar) {
    int i = blockIdx.x * 256 + threadIdx.x;          // 262144 threads
    if (i < Lsz * Lsz)   Wmbf[i] = f2bf(Wm[i]);
    if (i < Lsz * Dsz)   Wibf[i] = f2bf(Wi[i]);
    if (i < 2 * Bsz * Lsz) hbuf[i] = f2bf(0.0f);     // h_{-1} = 0 (ping-pong)
    if (i == 0) *bar = 0u;
}

// ---------------- phase 0b: x [B,T,D] fp32 -> time-major bf16 [T*B, D] ----------------
__global__ void __launch_bounds__(256)
xcvt_kernel(const float* __restrict__ x, __bf16* __restrict__ Xbf) {
    int o = blockIdx.x * 256 + threadIdx.x;          // 16,777,216 threads
    int d = o & (Dsz - 1);
    int r = o >> 8;                                  // r = t*16 + b
    int t = r >> 4;
    int b = r & 15;
    Xbf[o] = f2bf(x[((size_t)b * Tsz + t) * Dsz + d]);
}

// ---------------- phase 1: Z[TB, L] = X @ Wi^T  (bf16 WMMA, fp32 acc) ----------------
// grid (TB/16, 4), block 256 = 8 waves; wave w handles N-tile blockIdx.y*8 + w.
__global__ void __launch_bounds__(256)
gemm_xwi_kernel(const __bf16* __restrict__ Xbf, const __bf16* __restrict__ Wibf,
                __bf16* __restrict__ Z) {
    const int wave = threadIdx.x >> 5, lane = threadIdx.x & 31;
    const int lm = lane & 15, hi = lane >> 4;
    const int mbase = blockIdx.x * 16;
    const int nbase = blockIdx.y * 128 + wave * 16;

    // A fragment rows: x row (time-major); B fragment: Wi row n gives column n of Wi^T.
    const __bf16* arow = Xbf  + (size_t)(mbase + lm) * Dsz;
    const __bf16* brow = Wibf + (size_t)(nbase + lm) * Dsz;

    v8f c = {};
#pragma unroll
    for (int kc = 0; kc < Dsz / 32; ++kc) {          // 8 WMMAs, K = 256
        const int k0 = kc * 32 + hi * 8;             // ISA 16-bit A/B wave32 layout
        v8bf alo = *(const v8bf*)(arow + k0);
        v8bf ahi = *(const v8bf*)(arow + k0 + 16);
        v8bf blo = *(const v8bf*)(brow + k0);
        v8bf bhi = *(const v8bf*)(brow + k0 + 16);
        c = __builtin_amdgcn_wmma_f32_16x16x32_bf16(false, cat16(alo, ahi),
                                                    false, cat16(blo, bhi),
                                                    (short)0, c, false, false);
    }
    const int n = nbase + lm;
#pragma unroll
    for (int r = 0; r < 8; ++r)                      // C/D layout: lane%16 = N, vgpr r = M row
        Z[(size_t)(mbase + hi * 8 + r) * Lsz + n] = f2bf(c[r]);
}

// ---------------- phase 2: serial recurrence over T steps ----------------
// 16 blocks x 64 threads (2 waves). Block w owns columns [w*32, w*32+32) of L;
// its Wm slice (32x512 bf16) lives in LDS. h is exchanged via L2 (bf16, ping-pong).
__global__ void __launch_bounds__(64)
rnn_step_kernel(const __bf16* __restrict__ Z, const __bf16* __restrict__ Wmbf,
                const float* __restrict__ bm, float* __restrict__ hid,
                __bf16* __restrict__ hbuf, unsigned* __restrict__ bar) {
    __shared__ __bf16 lw[32 * LDS_PITCH];            // 33,280 B of the 320 KB WGP LDS

    const int wg = blockIdx.x, tid = threadIdx.x;
    for (int i = tid; i < 32 * Lsz; i += 64) {       // stage this block's Wm slice
        int c = i >> 9, k = i & (Lsz - 1);
        lw[c * LDS_PITCH + k] = Wmbf[(size_t)(wg * 32 + c) * Lsz + k];
    }
    __syncthreads();

    const int wave = tid >> 5, lane = tid & 31;
    const int lm = lane & 15, hi = lane >> 4;
    const int ncol = wg * 32 + wave * 16 + lm;       // global output column
    const int lcol = wave * 16 + lm;                 // column in LDS slice
    const float bias = bm[ncol];

    for (int t = 0; t < Tsz; ++t) {
        const __bf16* hprev = hbuf + (size_t)(t & 1) * (Bsz * Lsz);
        const __bf16* zt    = Z + (size_t)(t * Bsz) * Lsz;
        __builtin_prefetch(zt + Bsz * Lsz, 0, 1);    // global_prefetch next step's Z

        v8f c;
#pragma unroll
        for (int r = 0; r < 8; ++r)                  // C init = input projection + bias
            c[r] = bf2f(zt[(size_t)(hi * 8 + r) * Lsz + ncol]) + bias;

#pragma unroll 4
        for (int kc = 0; kc < Lsz / 32; ++kc) {      // 16 WMMAs, K = 512
            const int k0 = kc * 32 + hi * 8;
            v8bf alo = *(const v8bf*)(hprev + (size_t)lm * Lsz + k0);
            v8bf ahi = *(const v8bf*)(hprev + (size_t)lm * Lsz + k0 + 16);
            v8bf blo = *(const v8bf*)(&lw[lcol * LDS_PITCH + k0]);
            v8bf bhi = *(const v8bf*)(&lw[lcol * LDS_PITCH + k0 + 16]);
            c = __builtin_amdgcn_wmma_f32_16x16x32_bf16(false, cat16(alo, ahi),
                                                        false, cat16(blo, bhi),
                                                        (short)0, c, false, false);
        }

        __bf16* hnext = hbuf + (size_t)((t + 1) & 1) * (Bsz * Lsz);
        float*  hrow  = hid + (size_t)(t * Bsz) * Lsz;
#pragma unroll
        for (int r = 0; r < 8; ++r) {
            float s = 1.0f / (1.0f + __expf(-c[r])); // sigmoid
            hrow [(size_t)(hi * 8 + r) * Lsz + ncol] = s;        // fp32 hidden out
            hnext[(size_t)(hi * 8 + r) * Lsz + ncol] = f2bf(s);  // bf16 for next step
        }

        // ---- inter-workgroup step barrier ----
        __threadfence();
        __syncthreads();
        if (tid == 0)
            __hip_atomic_fetch_add(bar, 1u, __ATOMIC_RELEASE, __HIP_MEMORY_SCOPE_AGENT);
        __builtin_amdgcn_s_cluster_barrier();        // CDNA5 cluster sync (NOP un-clustered)
        if (tid == 0) {
            while (__hip_atomic_load(bar, __ATOMIC_ACQUIRE, __HIP_MEMORY_SCOPE_AGENT)
                   < (unsigned)(NWG * (t + 1)))
                __builtin_amdgcn_s_sleep(1);
        }
        __syncthreads();
        __threadfence();
    }
}

// ---------------- phase 3: outputs[t,b] = h[t,b,:] . Wo ----------------
__global__ void __launch_bounds__(256)
out_kernel(const float* __restrict__ hid, const float* __restrict__ Wo,
           float* __restrict__ out) {
    const int wave = threadIdx.x >> 5, lane = threadIdx.x & 31;
    const int row = blockIdx.x * 8 + wave;           // 0 .. TB-1
    float s = 0.0f;
    for (int i = lane; i < Lsz; i += 32)
        s += hid[(size_t)row * Lsz + i] * Wo[i];
#pragma unroll
    for (int off = 16; off > 0; off >>= 1)
        s += __shfl_down(s, off, 32);
    if (lane == 0) out[row] = s;
}

// ---------------- launcher ----------------
extern "C" void kernel_launch(void* const* d_in, const int* in_sizes, int n_in,
                              void* d_out, int out_size, void* d_ws, size_t ws_size,
                              hipStream_t stream) {
    const float* x  = (const float*)d_in[0];
    const float* Wi = (const float*)d_in[1];
    const float* Wm = (const float*)d_in[2];
    const float* bm = (const float*)d_in[3];
    const float* Wo = (const float*)d_in[4];

    float* out = (float*)d_out;                      // [T*B] outputs
    float* hid = out + TB;                           // [T*B*L] hidden states

    // workspace carve-up (all 256B-aligned by construction)
    char* ws = (char*)d_ws;
    const size_t ZB  = (size_t)TB * Lsz * 2;         // 64 MB
    const size_t XB  = (size_t)TB * Dsz * 2;         // 32 MB
    const size_t WIB = (size_t)Lsz * Dsz * 2;        // 256 KB
    const size_t WMB = (size_t)Lsz * Lsz * 2;        // 512 KB
    const size_t HB  = (size_t)2 * Bsz * Lsz * 2;    // 32 KB
    __bf16*  Z    = (__bf16*)(ws);
    __bf16*  Xbf  = (__bf16*)(ws + ZB);
    __bf16*  Wibf = (__bf16*)(ws + ZB + XB);
    __bf16*  Wmbf = (__bf16*)(ws + ZB + XB + WIB);
    __bf16*  hbuf = (__bf16*)(ws + ZB + XB + WIB + WMB);
    unsigned* bar = (unsigned*)(ws + ZB + XB + WIB + WMB + HB);

    prep_kernel<<<(Lsz * Lsz) / 256, 256, 0, stream>>>(Wi, Wm, Wibf, Wmbf, hbuf, bar);
    xcvt_kernel<<<((size_t)TB * Dsz) / 256, 256, 0, stream>>>(x, Xbf);
    gemm_xwi_kernel<<<dim3(TB / 16, Lsz / 128), 256, 0, stream>>>(Xbf, Wibf, Z);
    rnn_step_kernel<<<NWG, 64, 0, stream>>>(Z, Wmbf, bm, hid, hbuf, bar);
    out_kernel<<<TB / 8, 256, 0, stream>>>(hid, Wo, out);
}
